// Model_10634339025467
// MI455X (gfx1250) — matrix operand order repaired and checked by reading, not compiled, not run on hardware
//
#include <hip/hip_runtime.h>

typedef _Float16 h16;
typedef __attribute__((ext_vector_type(16))) _Float16 v16h;
typedef __attribute__((ext_vector_type(8)))  _Float16 v8h;
typedef __attribute__((ext_vector_type(8)))  float    v8f;

// payload type the async-LDS builtin expects: int __vector(4) (vector_size 16)
typedef int b128i __attribute__((vector_size(16)));
typedef __attribute__((address_space(1))) b128i* gp128;
typedef __attribute__((address_space(3))) b128i* lp128;

namespace {
constexpr int kS   = 2048;  // sequence length
constexpr int kD   = 64;    // head dim (QK)
constexpr int kE   = 64;    // V value dim
constexpr int kO   = 64;    // output dim
constexpr int kBH  = 64;    // B*H
constexpr int kWaves = 4;   // waves per workgroup -> 64 queries per WG
constexpr int kBc  = 64;    // keys per block iteration
constexpr size_t kQK = (size_t)kBH * kS * kD;  // elems per Q/K tensor
// softmax scale folded into log2 domain: log2(e)/sqrt(64), pre-applied to Q
constexpr float kScaleLog2 = 1.4426950408889634f / 8.0f;
}

__device__ __forceinline__ v8f wmma16(v16h a, v16h b, v8f c) {
  return __builtin_amdgcn_wmma_f32_16x16x32_f16(false, a, false, b, (short)0, c,
                                                false, false);
}

// ---- async global->LDS 16B copy (per-lane addresses), ASYNCcnt-tracked ----
__device__ __forceinline__ void async_cp16(const h16* gsrc, h16* ldst) {
#if __has_builtin(__builtin_amdgcn_global_load_async_to_lds_b128)
  __builtin_amdgcn_global_load_async_to_lds_b128(
      (gp128)(void*)gsrc, (lp128)(void*)ldst, 0, 0);
#else
  unsigned lofs = (unsigned)(size_t)(__attribute__((address_space(3))) void*)ldst;
  asm volatile("global_load_async_to_lds_b128 %0, %1, off"
               :: "v"(lofs), "v"(gsrc) : "memory");
#endif
}

__device__ __forceinline__ void wait_async0() {
#if __has_builtin(__builtin_amdgcn_s_wait_asynccnt)
  __builtin_amdgcn_s_wait_asynccnt(0);
#else
  asm volatile("s_wait_asynccnt 0x0" ::: "memory");
#endif
}

// A-operand fragment (16 x K=32 f16), d-chunk c, from 72-h16-stride LDS row.
// A layout: lanes 0-15: K = c*32 + 0..15;
//           lanes 16-31: K = c*32 + 16..31.
__device__ __forceinline__ v16h frag_a_row(const h16* __restrict__ row, int c, int hsel) {
  const v8h* rv = (const v8h*)row;
  v8h lo = rv[c * 4 + hsel * 2];
  v8h hi = rv[c * 4 + hsel * 2 + 1];
  return __builtin_shufflevector(lo, hi, 0, 1, 2, 3, 4, 5, 6, 7,
                                 8, 9, 10, 11, 12, 13, 14, 15);
}

// B-operand fragment (K=32 x 16 f16), k-chunk kc: lanes 0-15 K=kc*32+{0..7,16..23},
// lanes 16-31 K=kc*32+{8..15,24..31}.
__device__ __forceinline__ v16h frag_b_row(const h16* __restrict__ row, int kc, int hsel) {
  const v8h* rv = (const v8h*)row;
  v8h lo = rv[kc * 4 + hsel];
  v8h hi = rv[kc * 4 + hsel + 2];
  return __builtin_shufflevector(lo, hi, 0, 1, 2, 3, 4, 5, 6, 7,
                                 8, 9, 10, 11, 12, 13, 14, 15);
}

// B-operand fragment straight from global f16 (two aligned 16B loads).
__device__ __forceinline__ v16h frag_b_g(const h16* __restrict__ base, int idx) {
  const v8h* rv = (const v8h*)base;
  v8h lo = rv[idx];
  v8h hi = rv[idx + 1];
  return __builtin_shufflevector(lo, hi, 0, 1, 2, 3, 4, 5, 6, 7,
                                 8, 9, 10, 11, 12, 13, 14, 15);
}

// Pack 16 consecutive global floats (4 float4) into v16h.
__device__ __forceinline__ v16h pack16(const float4* __restrict__ p) {
  v16h r;
#pragma unroll
  for (int j = 0; j < 4; ++j) {
    float4 f = p[j];
    r[4 * j + 0] = (h16)f.x; r[4 * j + 1] = (h16)f.y;
    r[4 * j + 2] = (h16)f.z; r[4 * j + 3] = (h16)f.w;
  }
  return r;
}

__device__ __forceinline__ v8h pack8s(float4 a, float4 b, float s) {
  v8h r;
  r[0] = (h16)(a.x * s); r[1] = (h16)(a.y * s);
  r[2] = (h16)(a.z * s); r[3] = (h16)(a.w * s);
  r[4] = (h16)(b.x * s); r[5] = (h16)(b.y * s);
  r[6] = (h16)(b.z * s); r[7] = (h16)(b.w * s);
  return r;
}

// ===== prepass 1: Q (scaled) and K -> f16 ==================================
__global__ void convert_qk_kernel(const float* __restrict__ Q,
                                  const float* __restrict__ K,
                                  h16* __restrict__ q16, h16* __restrict__ k16) {
  const size_t i = ((size_t)blockIdx.x * blockDim.x + threadIdx.x) * 8;
  const float4* qs = (const float4*)(Q + i);
  const float4* ks = (const float4*)(K + i);
  *(v8h*)(q16 + i) = pack8s(qs[0], qs[1], kScaleLog2);
  *(v8h*)(k16 + i) = pack8s(ks[0], ks[1], 1.0f);
}

// ===== prepass 2: V -> f16, transposed per 64-key block: [bh][blk][e][k] ====
__global__ void transpose_v_kernel(const float* __restrict__ V,
                                   h16* __restrict__ vt16) {
  const int blk = blockIdx.x;      // 0..31
  const int bh  = blockIdx.y;      // 0..63
  const int t   = threadIdx.x;     // 256
  const int kk  = t & 63;
  const int e0  = (t >> 6) * 16;
  const float* src = V + ((size_t)bh * kS + blk * 64 + kk) * kE + e0;
  h16* dst = vt16 + ((size_t)(bh * 32 + blk)) * 4096 + kk;
#pragma unroll
  for (int i = 0; i < 16; ++i) dst[(size_t)(e0 + i) * 64] = (h16)src[i];
}

// ===== main: flash attention + projection ==================================
__launch_bounds__(128, 1)
__global__ void flash_attn_proj_kernel(const h16* __restrict__ Q16,
                                       const h16* __restrict__ K16,
                                       const h16* __restrict__ VT16,
                                       const float* __restrict__ W,
                                       const float* __restrict__ Bias,
                                       float* __restrict__ Out) {
  __shared__ __align__(16) h16 kT[2][64][72];       // K block, [key][d], dbuf
  __shared__ __align__(16) h16 vT[2][64][72];       // V^T block, [e][k], dbuf
  __shared__ __align__(16) h16 yS[kWaves][16][72];  // per-wave y tile, [q][e]

  const int bh   = blockIdx.y;
  const int q0   = blockIdx.x * (16 * kWaves);
  const int tid  = threadIdx.x;
  const int lane = tid & 31;
  const int wv   = tid >> 5;
  const int hsel = lane >> 4;
  const int l16  = lane & 15;

  const h16* Kg = K16  + (size_t)bh * kS * kD;                  // row-major f16
  const h16* Vg = VT16 + (size_t)bh * (kS / kBc) * (kBc * kE);  // blocked [e][k]

  // staging map: 512 16B-chunks per array, 4 per thread, padded LDS rows
  int srow[4], sch[4];
#pragma unroll
  for (int j = 0; j < 4; ++j) {
    const int c = tid + 128 * j;
    srow[j] = c >> 3;
    sch[j]  = (c & 7) * 8;
  }

  // ---- Q tile as B-operand (pre-scaled f16 in global) ---------------------
  const int qrow = q0 + wv * 16 + l16;
  const h16* qbase = Q16 + (size_t)bh * kS * kD + (size_t)qrow * kD;
  v16h qb0 = frag_b_g(qbase, 0 + hsel * 2);
  v16h qb1 = frag_b_g(qbase, 4 + hsel * 2);

  float m_run = -1e30f;
  float l_run = 0.0f;
  v8f o0 = {}, o1 = {}, o2 = {}, o3 = {};

  // prologue: stage block 0 into buffer 0
#pragma unroll
  for (int j = 0; j < 4; ++j) {
    async_cp16(Kg + (size_t)srow[j] * kD + sch[j], &kT[0][srow[j]][sch[j]]);
    async_cp16(Vg + (size_t)srow[j] * kBc + sch[j], &vT[0][srow[j]][sch[j]]);
  }
  wait_async0();
  __syncthreads();

  for (int kb = 0; kb < kS; kb += kBc) {
    const int cur = (kb >> 6) & 1;

    // ---- kick off async staging of the next block into the other buffer ---
    if (kb + kBc < kS) {
      const h16* kn = Kg + (size_t)(kb + kBc) * kD;
      const h16* vn = Vg + (size_t)((kb >> 6) + 1) * (kBc * kE);
      const int nxt = cur ^ 1;
#pragma unroll
      for (int j = 0; j < 4; ++j) {
        async_cp16(kn + (size_t)srow[j] * kD + sch[j], &kT[nxt][srow[j]][sch[j]]);
        async_cp16(vn + (size_t)srow[j] * kBc + sch[j], &vT[nxt][srow[j]][sch[j]]);
      }
    }

    // ---- S^T = K_block * Q^T : four 16x16 tiles ---------------------------
    v8f s0 = {}, s1 = {}, s2 = {}, s3 = {};
    {
      const h16* r0 = &kT[cur][ 0 + l16][0];
      const h16* r1 = &kT[cur][16 + l16][0];
      const h16* r2 = &kT[cur][32 + l16][0];
      const h16* r3 = &kT[cur][48 + l16][0];
      s0 = wmma16(frag_a_row(r0, 0, hsel), qb0, s0);
      s1 = wmma16(frag_a_row(r1, 0, hsel), qb0, s1);
      s2 = wmma16(frag_a_row(r2, 0, hsel), qb0, s2);
      s3 = wmma16(frag_a_row(r3, 0, hsel), qb0, s3);
      s0 = wmma16(frag_a_row(r0, 1, hsel), qb1, s0);
      s1 = wmma16(frag_a_row(r1, 1, hsel), qb1, s1);
      s2 = wmma16(frag_a_row(r2, 1, hsel), qb1, s2);
      s3 = wmma16(frag_a_row(r3, 1, hsel), qb1, s3);
    }

    // ---- online softmax (scores already in log2 domain) -------------------
    float mloc = -1e30f;
#pragma unroll
    for (int r = 0; r < 8; ++r)
      mloc = fmaxf(mloc, fmaxf(fmaxf(s0[r], s1[r]), fmaxf(s2[r], s3[r])));
    mloc = fmaxf(mloc, __shfl_xor(mloc, 16));
    const float mnew  = fmaxf(m_run, mloc);
    const float alpha = __builtin_amdgcn_exp2f(m_run - mnew);  // raw v_exp_f32

    float rsum = 0.0f;
    v16h pa0, pa1;  // P as A-operand: keys 0-31 and 32-63
#pragma unroll
    for (int r = 0; r < 8; ++r) {
      float e0 = __builtin_amdgcn_exp2f(s0[r] - mnew);
      float e1 = __builtin_amdgcn_exp2f(s1[r] - mnew);
      float e2 = __builtin_amdgcn_exp2f(s2[r] - mnew);
      float e3 = __builtin_amdgcn_exp2f(s3[r] - mnew);
      rsum += (e0 + e1) + (e2 + e3);
      pa0[r] = (h16)e0; pa0[r + 8] = (h16)e1;
      pa1[r] = (h16)e2; pa1[r + 8] = (h16)e3;
    }
    rsum += __shfl_xor(rsum, 16);
    l_run = l_run * alpha + rsum;
    m_run = mnew;

    // ---- rescale O only when some lane's max moved (wave-uniform skip) ----
    if (__ballot(alpha < 1.0f) != 0ull) {
      float ab[8];
#pragma unroll
      for (int r = 0; r < 8; ++r) ab[r] = __shfl(alpha, hsel * 8 + r);
#pragma unroll
      for (int r = 0; r < 8; ++r) {
        o0[r] *= ab[r]; o1[r] *= ab[r]; o2[r] *= ab[r]; o3[r] *= ab[r];
      }
    }

    // ---- O += P * V : 4 e-tiles x 2 k-chunks ------------------------------
    {
      const h16* v0 = &vT[cur][ 0 + l16][0];
      const h16* v1 = &vT[cur][16 + l16][0];
      const h16* v2 = &vT[cur][32 + l16][0];
      const h16* v3 = &vT[cur][48 + l16][0];
      o0 = wmma16(pa0, frag_b_row(v0, 0, hsel), o0);
      o1 = wmma16(pa0, frag_b_row(v1, 0, hsel), o1);
      o2 = wmma16(pa0, frag_b_row(v2, 0, hsel), o2);
      o3 = wmma16(pa0, frag_b_row(v3, 0, hsel), o3);
      o0 = wmma16(pa1, frag_b_row(v0, 1, hsel), o0);
      o1 = wmma16(pa1, frag_b_row(v1, 1, hsel), o1);
      o2 = wmma16(pa1, frag_b_row(v2, 1, hsel), o2);
      o3 = wmma16(pa1, frag_b_row(v3, 1, hsel), o3);
    }

    // my async copies done + everyone done reading `cur`
    wait_async0();
    __syncthreads();
  }

  // ---- finalize y = O / l, stage per-wave tile to LDS ---------------------
  const float invl = __builtin_amdgcn_rcpf(l_run);
  float ib[8];
#pragma unroll
  for (int r = 0; r < 8; ++r) ib[r] = __shfl(invl, hsel * 8 + r);
#pragma unroll
  for (int r = 0; r < 8; ++r) {
    const int qq = hsel * 8 + r;
    yS[wv][qq][0 * 16 + l16] = (h16)(o0[r] * ib[r]);
    yS[wv][qq][1 * 16 + l16] = (h16)(o1[r] * ib[r]);
    yS[wv][qq][2 * 16 + l16] = (h16)(o2[r] * ib[r]);
    yS[wv][qq][3 * 16 + l16] = (h16)(o3[r] * ib[r]);
  }
  // same-wave LDS write->read: DScnt keeps these ordered, no barrier needed.

  // ---- output projection: out = y * W^T + b (WMMA) ------------------------
  v16h ya0 = frag_a_row(&yS[wv][l16][0], 0, hsel);
  v16h ya1 = frag_a_row(&yS[wv][l16][0], 1, hsel);

#pragma unroll
  for (int ot = 0; ot < 4; ++ot) {
    const int oc = ot * 16 + l16;
    const float4* wp = (const float4*)(W + (size_t)oc * kE);
    v16h wb0 = pack16(wp + 0 * 8 + hsel * 4);
    v16h wb1 = pack16(wp + 1 * 8 + hsel * 4);
    v8f acc = {};
    acc = wmma16(ya0, wb0, acc);
    acc = wmma16(ya1, wb1, acc);
    const float bias = Bias[oc];
#pragma unroll
    for (int r = 0; r < 8; ++r) {
      const int qq = q0 + wv * 16 + hsel * 8 + r;
      Out[((size_t)bh * kS + qq) * kO + oc] = acc[r] + bias;
    }
  }
}

extern "C" void kernel_launch(void* const* d_in, const int* in_sizes, int n_in,
                              void* d_out, int out_size, void* d_ws, size_t ws_size,
                              hipStream_t stream) {
  (void)in_sizes; (void)n_in; (void)ws_size; (void)out_size;
  const float* q     = (const float*)d_in[0];
  const float* k     = (const float*)d_in[1];
  const float* v     = (const float*)d_in[2];
  const float* w_out = (const float*)d_in[3];
  const float* b_out = (const float*)d_in[4];
  float* out = (float*)d_out;

  // workspace layout: Q16 | K16 | VT16 (each kQK f16 = 16 MiB; total 48 MiB)
  h16* q16  = (h16*)d_ws;
  h16* k16  = q16 + kQK;
  h16* vt16 = k16 + kQK;

  // prepass: f32 -> f16 (Q pre-scaled), V transposed per 64-key block
  convert_qk_kernel<<<dim3((unsigned)(kQK / 8 / 256)), dim3(256), 0, stream>>>(
      q, k, q16, k16);
  transpose_v_kernel<<<dim3(kS / kBc, kBH), dim3(256), 0, stream>>>(v, vt16);

  // main fused flash-attention + projection
  dim3 grid(kS / (16 * kWaves), kBH);  // (32, 64)
  dim3 block(32 * kWaves);             // 128 threads = 4 waves
  flash_attn_proj_kernel<<<grid, block, 0, stream>>>(q16, k16, vt16,
                                                     w_out, b_out, out);
}
